// EnhancedHomoGNN_66374424592516
// MI455X (gfx1250) — matrix operand order repaired and checked by reading, not compiled, not run on hardware
//
#include <hip/hip_runtime.h>
#include <math.h>

// ---------------------------------------------------------------------------
// EnhancedHomoGNN forward for MI455X (gfx1250, wave32, WMMA)
// ---------------------------------------------------------------------------

#define HID   128
#define HEADS 8
#define CH    16
#define NGR   64      // number of graphs (G)
#define NEGS  0.2f    // leaky-relu slope

typedef __attribute__((ext_vector_type(16))) __bf16 v16bf16;
typedef __attribute__((ext_vector_type(8)))  float  v8f;
typedef __attribute__((ext_vector_type(4)))  unsigned int v4u;

// float -> bf16 (round to nearest even)
__device__ inline unsigned short f2bf(float f) {
    unsigned int u = __float_as_uint(f);
    unsigned int r = u + 0x7FFFu + ((u >> 16) & 1u);
    return (unsigned short)(r >> 16);
}

// float atomic max via sign-split integer atomics (portable & correct)
__device__ inline void atomicMaxF(float* addr, float val) {
    if (val >= 0.f) atomicMax((int*)addr, __float_as_int(val));
    else            atomicMin((unsigned int*)addr, __float_as_uint(val));
}

// ---------------------------------------------------------------------------
// WMMA GEMM: out[n,0:128] = act(A[n,0:K] @ W[K,128] + bias (+ out if ACC))
// Block: 256 thr = 8 waves; block covers 16 rows x 128 cols; wave -> 16x16 tile.
// A staged row-major, W staged TRANSPOSED ([col][k]) in LDS as bf16 so each
// WMMA fragment is two contiguous ds_load_b128 per operand.
// ---------------------------------------------------------------------------
template <int K, int ACT, bool ACC>
__global__ __launch_bounds__(256) void gemm16(const float* __restrict__ A,
                                              const float* __restrict__ W,
                                              const float* __restrict__ bias,
                                              float* __restrict__ out, int nrows) {
    constexpr int KP = K + 8;                 // +16B row pad: keeps 16B align, staggers banks
    __shared__ unsigned short sWt[HID * KP];  // W^T as bf16: [col][k]
    __shared__ unsigned short sA[16 * KP];    // A tile as bf16: [row][k]
    __shared__ float          sB[HID];

    const int t    = threadIdx.x;
    const int row0 = blockIdx.x << 4;

    // warm GL2 with the A tile while W staging streams (global_prefetch_b8)
    __builtin_prefetch(&A[(size_t)(row0 + (t & 15)) * K], 0, 1);

    for (int i = t; i < K * HID; i += 256) {           // coalesced global read
        int k = i >> 7, col = i & 127;
        sWt[col * KP + k] = f2bf(W[i]);                // transposed LDS write
    }
    for (int i = t; i < 16 * K; i += 256) {
        int r = i / K, k = i - r * K;
        int gr = row0 + r;
        float v = (gr < nrows) ? A[(size_t)gr * K + k] : 0.f;
        sA[r * KP + k] = f2bf(v);
    }
    if (t < HID) sB[t] = bias ? bias[t] : 0.f;
    __syncthreads();

    const int wave  = t >> 5;
    const int lane  = t & 31;
    const int khalf = lane >> 4;   // which K-half this lane holds
    const int lr    = lane & 15;   // A: row in tile / B,D: column in tile
    const int col   = (wave << 4) + lr;

    // bias folded into accumulator init: every C row of this lane shares `col`
    const float bv = sB[col];
    v8f acc = {bv, bv, bv, bv, bv, bv, bv, bv};

    union Frag { v4u q[2]; v16bf16 v; };

#pragma unroll
    for (int kb = 0; kb < K; kb += 32) {
        Frag fa, fb;
        // A 16x32 bf16 striping (ISA 7.12.2): elems 0..7 -> k=kb+8*khalf+e,
        // elems 8..15 -> +16; both halves contiguous -> 2x ds_load_b128
        const int ab = lr * KP + kb + 8 * khalf;
        fa.q[0] = *(const v4u*)&sA[ab];
        fa.q[1] = *(const v4u*)&sA[ab + 16];
        // B 32x16 bf16 striping (same as A): elems 0..7 -> k=kb+8*khalf+e,
        // elems 8..15 -> +16; fixed col in transposed LDS -> 2x ds_load_b128
        const int bb = col * KP + kb + 8 * khalf;
        fb.q[0] = *(const v4u*)&sWt[bb];
        fb.q[1] = *(const v4u*)&sWt[bb + 16];
        acc = __builtin_amdgcn_wmma_f32_16x16x32_bf16(false, fa.v, false, fb.v,
                                                      (short)0, acc, false, false);
    }

    // D layout: lane = col, VGPR r -> row r (+8 for upper lane half)
    const int rb = (lane >= 16) ? 8 : 0;
    float* po = out + (size_t)(row0 + rb) * HID + col;
    if (row0 + 16 <= nrows) {                 // full tile: unguarded stores
#pragma unroll
        for (int r = 0; r < 8; ++r) {
            float v = acc[r];
            if (ACC) v += po[(size_t)r * HID];
            if (ACT == 1) v = fmaxf(v, 0.f);
            po[(size_t)r * HID] = v;
        }
    } else {
#pragma unroll
        for (int r = 0; r < 8; ++r) {
            if (row0 + rb + r < nrows) {
                float v = acc[r];
                if (ACC) v += po[(size_t)r * HID];
                if (ACT == 1) v = fmaxf(v, 0.f);
                po[(size_t)r * HID] = v;
            }
        }
    }
}

// ---------------------------------------------------------------------------
// Elementwise / row-wise helpers
// ---------------------------------------------------------------------------
__global__ void fill_kernel(float* p, float v, long long n) {
    long long i = (long long)blockIdx.x * blockDim.x + threadIdx.x;
    if (i < n) p[i] = v;
}

__global__ void wsum_kernel(const float* Wi, const float* Wr, const float* bi,
                            const float* br, float* wsum, float* bsum) {
    int i = blockIdx.x * blockDim.x + threadIdx.x;
    if (i < 64 * HID) wsum[i] = Wi[i] + Wr[i];
    if (i < HID)      bsum[i] = bi[i] + br[i];
}

// LayerNorm + activation (ACT: 2 = ELU, 3 = LeakyReLU(0.2)); wave per row
template <int ACT>
__global__ __launch_bounds__(256) void ln_act_kernel(const float* __restrict__ in,
                                                     const float* __restrict__ g,
                                                     const float* __restrict__ b,
                                                     float* __restrict__ outp, int n) {
    int wave = threadIdx.x >> 5, lane = threadIdx.x & 31;
    int row = blockIdx.x * 8 + wave;
    if (row >= n) return;
    float v[4];
    float s = 0.f;
#pragma unroll
    for (int i = 0; i < 4; i++) {
        v[i] = in[(size_t)row * HID + lane * 4 + i];
        s += v[i];
    }
#pragma unroll
    for (int off = 16; off > 0; off >>= 1) s += __shfl_xor(s, off, 32);
    float mu = s * (1.f / HID);
    float var = 0.f;
#pragma unroll
    for (int i = 0; i < 4; i++) { float d = v[i] - mu; var += d * d; }
#pragma unroll
    for (int off = 16; off > 0; off >>= 1) var += __shfl_xor(var, off, 32);
    var *= (1.f / HID);
    float rs = rsqrtf(var + 1e-5f);
#pragma unroll
    for (int i = 0; i < 4; i++) {
        int c = lane * 4 + i;
        float y = (v[i] - mu) * rs * g[c] + b[c];
        if (ACT == 2) y = (y > 0.f) ? y : expm1f(y);
        else          y = (y >= 0.f) ? y : NEGS * y;
        outp[(size_t)row * HID + c] = y;
    }
}

// row-wise L2 normalize in place; wave per row
__global__ __launch_bounds__(256) void l2norm_kernel(float* p, int n) {
    int wave = threadIdx.x >> 5, lane = threadIdx.x & 31;
    int row = blockIdx.x * 8 + wave;
    if (row >= n) return;
    float v[4];
    float s = 0.f;
#pragma unroll
    for (int i = 0; i < 4; i++) {
        v[i] = p[(size_t)row * HID + lane * 4 + i];
        s += v[i] * v[i];
    }
#pragma unroll
    for (int off = 16; off > 0; off >>= 1) s += __shfl_xor(s, off, 32);
    float inv = 1.f / fmaxf(sqrtf(s), 1e-12f);
#pragma unroll
    for (int i = 0; i < 4; i++) p[(size_t)row * HID + lane * 4 + i] = v[i] * inv;
}

// ---------------------------------------------------------------------------
// GATv2 edge phase (self-loops appended virtually at indices >= E)
// ---------------------------------------------------------------------------
__global__ void gat_init_kernel(float* conv, const float* bias, float* m,
                                float* den, int n) {
    long long i = (long long)blockIdx.x * blockDim.x + threadIdx.x;
    if (i < (long long)n * HID) conv[i] = bias[i & (HID - 1)];
    if (i < (long long)n * HEADS) { m[i] = __uint_as_float(0xFF800000u); den[i] = 0.f; }
}

__global__ void gat_score_kernel(const float* __restrict__ xl, const float* __restrict__ xr,
                                 const float* __restrict__ att, const int* __restrict__ src,
                                 const int* __restrict__ dst, int ne, int n,
                                 float* __restrict__ esc, float* __restrict__ m) {
    long long i = (long long)blockIdx.x * blockDim.x + threadIdx.x;
    long long tot = ((long long)ne + n) * HEADS;
    if (i >= tot) return;
    int hh = (int)(i & (HEADS - 1));
    long long e = i >> 3;
    int s, d;
    if (e < ne) { s = src[e]; d = dst[e]; } else { s = d = (int)(e - ne); }
    const float* a  = att + hh * CH;
    const float* pl = xl + (size_t)s * HID + hh * CH;
    const float* pr = xr + (size_t)d * HID + hh * CH;
    float sc = 0.f;
#pragma unroll
    for (int c = 0; c < CH; c++) {
        float z = pl[c] + pr[c];
        z = (z >= 0.f) ? z : NEGS * z;
        sc += z * a[c];
    }
    esc[i] = sc;
    atomicMaxF(&m[(size_t)d * HEADS + hh], sc);
}

__global__ void gat_weight_kernel(float* __restrict__ esc, const float* __restrict__ m,
                                  float* __restrict__ den, const int* __restrict__ dst,
                                  int ne, int n) {
    long long i = (long long)blockIdx.x * blockDim.x + threadIdx.x;
    long long tot = ((long long)ne + n) * HEADS;
    if (i >= tot) return;
    int hh = (int)(i & (HEADS - 1));
    long long e = i >> 3;
    int d = (e < ne) ? dst[e] : (int)(e - ne);
    float w = expf(esc[i] - m[(size_t)d * HEADS + hh]);
    esc[i] = w;
    atomicAdd(&den[(size_t)d * HEADS + hh], w);
}

__global__ void gat_agg_kernel(const float* __restrict__ esc, const float* __restrict__ den,
                               const float* __restrict__ xl, const int* __restrict__ src,
                               const int* __restrict__ dst, int ne, int n,
                               float* __restrict__ conv) {
    long long i = (long long)blockIdx.x * blockDim.x + threadIdx.x;
    long long tot = ((long long)ne + n) * HEADS;
    if (i >= tot) return;
    int hh = (int)(i & (HEADS - 1));
    long long e = i >> 3;
    int s, d;
    if (e < ne) { s = src[e]; d = dst[e]; } else { s = d = (int)(e - ne); }
    float alpha = esc[i] / (den[(size_t)d * HEADS + hh] + 1e-16f);
    const float* pl = xl + (size_t)s * HID + hh * CH;
    float* po = conv + (size_t)d * HID + hh * CH;
#pragma unroll
    for (int c = 0; c < CH; c++) atomicAdd(&po[c], alpha * pl[c]);
}

// ---------------------------------------------------------------------------
// SAGE edge phase (no self-loops)
// ---------------------------------------------------------------------------
__global__ void sage_agg_kernel(const float* __restrict__ xp, const int* __restrict__ src,
                                const int* __restrict__ dst, int ne,
                                float* __restrict__ agg, float* __restrict__ cnt) {
    long long i = (long long)blockIdx.x * blockDim.x + threadIdx.x;
    long long tot = (long long)ne * 4;
    if (i >= tot) return;
    int q = (int)(i & 3);
    long long e = i >> 2;
    int s = src[e], d = dst[e];
    const float* p = xp + (size_t)s * HID + q * 32;
    float* o = agg + (size_t)d * HID + q * 32;
#pragma unroll
    for (int c = 0; c < 32; c++) atomicAdd(&o[c], p[c]);
    if (q == 0) atomicAdd(&cnt[d], 1.f);
}

__global__ void sage_mean_kernel(float* agg, const float* cnt, int n) {
    long long i = (long long)blockIdx.x * blockDim.x + threadIdx.x;
    if (i < (long long)n * HID) {
        int row = (int)(i >> 7);
        agg[i] = agg[i] / fmaxf(cnt[row], 1.f);
    }
}

// ---------------------------------------------------------------------------
// JK-max + sigmoid gate + global mean-pool accumulation; wave per row
// ---------------------------------------------------------------------------
__global__ __launch_bounds__(256) void jk_pool_kernel(const float* x0, const float* x1,
                                                      const float* x2, const float* x3,
                                                      const float* Wap, const float* bap,
                                                      const int* batch, float* pooled,
                                                      float* cntg, int n) {
    int wave = threadIdx.x >> 5, lane = threadIdx.x & 31;
    int row = blockIdx.x * 8 + wave;
    if (row >= n) return;
    float hv[4];
    float dot = 0.f;
#pragma unroll
    for (int i = 0; i < 4; i++) {
        int c = lane * 4 + i;
        size_t idx = (size_t)row * HID + c;
        float v = fmaxf(fmaxf(x0[idx], x1[idx]), fmaxf(x2[idx], x3[idx]));
        hv[i] = v;
        dot += v * Wap[c];
    }
#pragma unroll
    for (int off = 16; off > 0; off >>= 1) dot += __shfl_xor(dot, off, 32);
    float gate = 1.f / (1.f + expf(-(dot + bap[0])));
    int b = batch[row];
#pragma unroll
    for (int i = 0; i < 4; i++) {
        int c = lane * 4 + i;
        atomicAdd(&pooled[(size_t)b * HID + c], hv[i] * gate);
    }
    if (lane == 0) atomicAdd(&cntg[b], 1.f);
}

// ---------------------------------------------------------------------------
// Classifier head: mean -> prelu(W1) -> prelu(W2) -> W3 ; single block
// ---------------------------------------------------------------------------
__global__ __launch_bounds__(256) void classifier_kernel(
        const float* pooled, const float* cntg,
        const float* Wc1, const float* bc1, const float* a1,
        const float* Wc2, const float* bc2, const float* a2,
        const float* Wc3, const float* bc3, float* out) {
    __shared__ float sp[NGR * HID];
    __shared__ float s1[NGR * HID];
    int t = threadIdx.x;
    for (int i = t; i < NGR * HID; i += 256) {
        int g = i >> 7;
        sp[i] = pooled[i] / fmaxf(cntg[g], 1.f);
    }
    __syncthreads();
    float A1 = a1[0], A2 = a2[0];
    for (int i = t; i < NGR * HID; i += 256) {
        int g = i >> 7, c = i & 127;
        float acc = bc1[c];
        for (int k = 0; k < HID; k++) acc += sp[g * HID + k] * Wc1[k * HID + c];
        s1[i] = (acc >= 0.f) ? acc : A1 * acc;
    }
    __syncthreads();
    for (int i = t; i < NGR * 64; i += 256) {
        int g = i >> 6, c = i & 63;
        float acc = bc2[c];
        for (int k = 0; k < HID; k++) acc += s1[g * HID + k] * Wc2[k * 64 + c];
        sp[i] = (acc >= 0.f) ? acc : A2 * acc;   // reuse sp as z2 [64][64]
    }
    __syncthreads();
    for (int i = t; i < NGR * 2; i += 256) {
        int g = i >> 1, c = i & 1;
        float acc = bc3[c];
        for (int k = 0; k < 64; k++) acc += sp[g * 64 + k] * Wc3[k * 2 + c];
        out[i] = acc;
    }
}

// ---------------------------------------------------------------------------
static inline int cdivll(long long a, long long b) { return (int)((a + b - 1) / b); }

extern "C" void kernel_launch(void* const* d_in, const int* in_sizes, int n_in,
                              void* d_out, int out_size, void* d_ws, size_t ws_size,
                              hipStream_t stream) {
    const float* x      = (const float*)d_in[0];
    const int*   ei     = (const int*)d_in[1];
    const int*   batch  = (const int*)d_in[2];
    const float* W_in   = (const float*)d_in[3];
    const float* b_in   = (const float*)d_in[4];
    const float* W_res  = (const float*)d_in[5];
    const float* b_res  = (const float*)d_in[6];
    const float* g0_Wl  = (const float*)d_in[7];
    const float* g0_bl  = (const float*)d_in[8];
    const float* g0_Wr  = (const float*)d_in[9];
    const float* g0_br  = (const float*)d_in[10];
    const float* g0_att = (const float*)d_in[11];
    const float* g0_bias= (const float*)d_in[12];
    const float* g2_Wl  = (const float*)d_in[13];
    const float* g2_bl  = (const float*)d_in[14];
    const float* g2_Wr  = (const float*)d_in[15];
    const float* g2_br  = (const float*)d_in[16];
    const float* g2_att = (const float*)d_in[17];
    const float* g2_bias= (const float*)d_in[18];
    const float* s1_Wp  = (const float*)d_in[19];
    const float* s1_bp  = (const float*)d_in[20];
    const float* s1_Wl  = (const float*)d_in[21];
    const float* s1_bl  = (const float*)d_in[22];
    const float* s1_Wr  = (const float*)d_in[23];
    const float* s3_Wp  = (const float*)d_in[24];
    const float* s3_bp  = (const float*)d_in[25];
    const float* s3_Wl  = (const float*)d_in[26];
    const float* s3_bl  = (const float*)d_in[27];
    const float* s3_Wr  = (const float*)d_in[28];
    const float* ln_g   = (const float*)d_in[29];
    const float* ln_b   = (const float*)d_in[30];
    const float* Wap    = (const float*)d_in[31];
    const float* bap    = (const float*)d_in[32];
    const float* Wc1    = (const float*)d_in[33];
    const float* bc1    = (const float*)d_in[34];
    const float* a1     = (const float*)d_in[35];
    const float* Wc2    = (const float*)d_in[36];
    const float* bc2    = (const float*)d_in[37];
    const float* a2     = (const float*)d_in[38];
    const float* Wc3    = (const float*)d_in[39];
    const float* bc3    = (const float*)d_in[40];

    const int N = in_sizes[0] / 64;
    const int E = in_sizes[1] / 2;
    const int* src = ei;
    const int* dst = ei + E;

    // workspace layout (floats)
    float* ws = (float*)d_ws;
    const size_t NH = (size_t)N * HID;
    float* h    = ws;
    float* xl   = h + NH;
    float* xr   = xl + NH;
    float* conv = xr + NH;
    float* xs0  = conv + NH;
    float* xs1  = xs0 + NH;
    float* xs2  = xs1 + NH;
    float* xs3  = xs2 + NH;
    float* esc  = xs3 + NH;                 // (E+N)*HEADS
    const size_t EH = (size_t)(E + N) * HEADS;
    float* mbf  = esc + EH;                 // N*HEADS
    float* den  = mbf + (size_t)N * HEADS;  // N*HEADS
    float* cnt  = den + (size_t)N * HEADS;  // N
    float* pooled = cnt + N;                // NGR*HID
    float* cntg = pooled + NGR * HID;       // NGR
    float* wsum = cntg + NGR;               // 64*HID
    float* bsum = wsum + 64 * HID;          // HID

    const int gB = cdivll(N, 16);           // GEMM blocks (16 rows each)
    const int rB = cdivll(N, 8);            // wave-per-row blocks

    // h = x @ (W_in+W_res) + (b_in+b_res)
    wsum_kernel<<<cdivll(64 * HID, 256), 256, 0, stream>>>(W_in, W_res, b_in, b_res, wsum, bsum);
    gemm16<64, 0, false><<<gB, 256, 0, stream>>>(x, wsum, bsum, h, N);

    auto gat = [&](const float* hin, const float* Wl, const float* bl,
                   const float* Wr, const float* br, const float* att,
                   const float* bias, const float* lng, const float* lnb, float* xout) {
        gemm16<128, 0, false><<<gB, 256, 0, stream>>>(hin, Wl, bl, xl, N);
        gemm16<128, 0, false><<<gB, 256, 0, stream>>>(hin, Wr, br, xr, N);
        gat_init_kernel<<<cdivll((long long)N * HID, 256), 256, 0, stream>>>(conv, bias, mbf, den, N);
        long long tot = ((long long)E + N) * HEADS;
        gat_score_kernel<<<cdivll(tot, 256), 256, 0, stream>>>(xl, xr, att, src, dst, E, N, esc, mbf);
        gat_weight_kernel<<<cdivll(tot, 256), 256, 0, stream>>>(esc, mbf, den, dst, E, N);
        gat_agg_kernel<<<cdivll(tot, 256), 256, 0, stream>>>(esc, den, xl, src, dst, E, N, conv);
        ln_act_kernel<2><<<rB, 256, 0, stream>>>(conv, lng, lnb, xout, N);  // ELU
    };

    auto sage = [&](const float* hin, const float* Wp, const float* bp,
                    const float* Wl, const float* bl, const float* Wr,
                    const float* lng, const float* lnb, float* xout) {
        gemm16<128, 1, false><<<gB, 256, 0, stream>>>(hin, Wp, bp, xl, N);      // xp = relu
        fill_kernel<<<cdivll((long long)NH, 256), 256, 0, stream>>>(xr, 0.f, (long long)NH);
        fill_kernel<<<cdivll(N, 256), 256, 0, stream>>>(cnt, 0.f, N);
        sage_agg_kernel<<<cdivll((long long)E * 4, 256), 256, 0, stream>>>(xl, src, dst, E, xr, cnt);
        sage_mean_kernel<<<cdivll((long long)NH, 256), 256, 0, stream>>>(xr, cnt, N);
        gemm16<128, 0, false><<<gB, 256, 0, stream>>>(xr, Wl, bl, conv, N);     // mean@Wl+bl
        gemm16<128, 0, true ><<<gB, 256, 0, stream>>>(hin, Wr, nullptr, conv, N); // += h@Wr
        l2norm_kernel<<<rB, 256, 0, stream>>>(conv, N);
        ln_act_kernel<3><<<rB, 256, 0, stream>>>(conv, lng, lnb, xout, N);      // leaky
    };

    gat (h,   g0_Wl, g0_bl, g0_Wr, g0_br, g0_att, g0_bias, ln_g + 0 * HID, ln_b + 0 * HID, xs0);
    sage(xs0, s1_Wp, s1_bp, s1_Wl, s1_bl, s1_Wr,           ln_g + 1 * HID, ln_b + 1 * HID, xs1);
    gat (xs1, g2_Wl, g2_bl, g2_Wr, g2_br, g2_att, g2_bias, ln_g + 2 * HID, ln_b + 2 * HID, xs2);
    sage(xs2, s3_Wp, s3_bp, s3_Wl, s3_bl, s3_Wr,           ln_g + 3 * HID, ln_b + 3 * HID, xs3);

    fill_kernel<<<cdivll(NGR * HID, 256), 256, 0, stream>>>(pooled, 0.f, NGR * HID);
    fill_kernel<<<1, 64, 0, stream>>>(cntg, 0.f, NGR);
    jk_pool_kernel<<<rB, 256, 0, stream>>>(xs0, xs1, xs2, xs3, Wap, bap, batch, pooled, cntg, N);

    classifier_kernel<<<1, 256, 0, stream>>>(pooled, cntg, Wc1, bc1, a1,
                                             Wc2, bc2, a2, Wc3, bc3, (float*)d_out);
}
